// NodeVectorOutputHead_68298569941526
// MI455X (gfx1250) — matrix operand order repaired and checked by reading, not compiled, not run on hardware
//
#include <hip/hip_runtime.h>
#include <hip/hip_bf16.h>

// CDNA5 / gfx1250: wave32, WMMA bf16 16x16x32, async global->LDS staging.
typedef __attribute__((ext_vector_type(16))) __bf16 v16bf;
typedef __attribute__((ext_vector_type(8)))  float  v8f;
typedef __attribute__((ext_vector_type(4)))  float  v4f;

#define D_MODEL 256
#define M_TILE  128    // rows per workgroup (8 waves x 16 rows)
#define K_STEP  32
#define N_TILES 16
#define K_STEPS 8      // 256 / 32
#define SLAB_BYTES 16384   // 16 tiles * 32 lanes * 16 bf16 * 2B = 16 KB per (hi|lo) plane

// ---- gfx1250-specific helpers ------------------------------------------------

// Low 32 bits of a flat pointer into __shared__ == LDS byte offset (aperture form).
__device__ __forceinline__ unsigned lds_addr32(const void* p) {
  return (unsigned)(unsigned long long)p;
}

// GLOBAL_LOAD_ASYNC_TO_LDS_B128 (tracked by ASYNCcnt), GV addressing mode.
__device__ __forceinline__ void async_ld_b128(unsigned lds, const void* g) {
  asm volatile("global_load_async_to_lds_b128 %0, %1, off"
               :: "v"(lds), "v"(g) : "memory");
}

__device__ __forceinline__ void wait_async_zero() {
  asm volatile("s_wait_asynccnt 0" ::: "memory");
}

// ---- W1 repack: fp32 -> bf16 hi/lo planes in per-lane WMMA B-fragment order ---
// B (32x16 bf16, v16bf/lane): lane l holds column N = l&15, K = (l>>4)*16 + i.
__global__ __launch_bounds__(256) void nvoh_pack_w1(const float* __restrict__ W1,
                                                    void* __restrict__ bhi_,
                                                    void* __restrict__ blo_) {
  __bf16* Bhi = (__bf16*)bhi_;
  __bf16* Blo = (__bf16*)blo_;
  int id = blockIdx.x * 256 + threadIdx.x;   // id = k*256 + n (W1 row-major [K][N])
  int k = id >> 8, n = id & 255;
  int s = k >> 5, ko = k & 31;               // k-slab, k within slab
  int t = n >> 4, c = n & 15;                // n-tile, column within tile
  int l = c | ((ko >> 4) << 4);              // lane
  int i = ko & 15;                           // element within lane fragment
  float w = W1[id];
  __bf16 h = (__bf16)w;
  int dst = ((s * N_TILES + t) * 32 + l) * 16 + i;
  Bhi[dst] = h;
  Blo[dst] = (__bf16)(w - (float)h);
}

// ---- Fused: GEMM(bf16x3 split) -> +b1 -> SiLU -> dot W2 -> s*V_st -> scatter --
__global__ __launch_bounds__(256) void nvoh_main(
    const float* __restrict__ forces, const float* __restrict__ V_st,
    const float* __restrict__ b1, const float* __restrict__ W2,
    const float* __restrict__ b2, const int* __restrict__ idx_t,
    const void* __restrict__ bhi_, const void* __restrict__ blo_,
    float* __restrict__ out, int E)
{
  __shared__ __align__(32) char ldsB[2][2][SLAB_BYTES];  // [buf][hi|lo][16KB]
  const char* Bhi = (const char*)bhi_;
  const char* Blo = (const char*)blo_;

  const int tid   = threadIdx.x;
  const int lane  = tid & 31;
  const int wave  = tid >> 5;
  const int sub   = lane & 15;
  const int khalf = lane >> 4;

  const int waveRow = blockIdx.x * M_TILE + wave * 16;

  // A-fragment row fed by this lane (clamped; duplicates are never written back)
  int arow = waveRow + sub;
  if (arow > E - 1) arow = E - 1;
  const float* rowPtr = forces + (size_t)arow * D_MODEL;

  v8f acc[N_TILES];
#pragma unroll
  for (int t = 0; t < N_TILES; ++t) acc[t] = (v8f){0,0,0,0,0,0,0,0};

  // Stage slab 0 (async DMA into LDS buffer 0)
#pragma unroll
  for (int it = 0; it < 4; ++it) {
    int off = (it * 256 + tid) * 16;
    async_ld_b128(lds_addr32(&ldsB[0][0][0]) + off, Bhi + off);
    async_ld_b128(lds_addr32(&ldsB[0][1][0]) + off, Blo + off);
  }
  wait_async_zero();
  __syncthreads();

  int buf = 0;
  for (int s = 0; s < K_STEPS; ++s) {
    // Prefetch next k-slab into the other buffer while we compute this one.
    if (s + 1 < K_STEPS) {
      const char* gH = Bhi + (size_t)(s + 1) * SLAB_BYTES;
      const char* gL = Blo + (size_t)(s + 1) * SLAB_BYTES;
      unsigned dH = lds_addr32(&ldsB[buf ^ 1][0][0]);
      unsigned dL = lds_addr32(&ldsB[buf ^ 1][1][0]);
#pragma unroll
      for (int it = 0; it < 4; ++it) {
        int off = (it * 256 + tid) * 16;
        async_ld_b128(dH + off, gH + off);
        async_ld_b128(dL + off, gL + off);
      }
    }

    // A fragment (16x32 bf16): lane l -> row M=l&15, K = k0..k0+15.
    const int k0 = s * K_STEP + khalf * 16;
    v4f f0 = __builtin_nontemporal_load((const v4f*)(rowPtr + k0));
    v4f f1 = __builtin_nontemporal_load((const v4f*)(rowPtr + k0 + 4));
    v4f f2 = __builtin_nontemporal_load((const v4f*)(rowPtr + k0 + 8));
    v4f f3 = __builtin_nontemporal_load((const v4f*)(rowPtr + k0 + 12));
    float av[16] = {f0.x, f0.y, f0.z, f0.w, f1.x, f1.y, f1.z, f1.w,
                    f2.x, f2.y, f2.z, f2.w, f3.x, f3.y, f3.z, f3.w};
    v16bf ah, al;   // fp32 = hi(bf16) + lo(bf16) split for ~fp32 product accuracy
#pragma unroll
    for (int i = 0; i < 16; ++i) {
      __bf16 h = (__bf16)av[i];
      ah[i] = h;
      al[i] = (__bf16)(av[i] - (float)h);
    }

#pragma unroll
    for (int t = 0; t < N_TILES; ++t) {
      const v16bf bh = *(const v16bf*)&ldsB[buf][0][(t * 32 + lane) * 32];
      const v16bf bl = *(const v16bf*)&ldsB[buf][1][(t * 32 + lane) * 32];
      acc[t] = __builtin_amdgcn_wmma_f32_16x16x32_bf16(false, ah, false, bh,
                                                       (short)0, acc[t], false, false);
      acc[t] = __builtin_amdgcn_wmma_f32_16x16x32_bf16(false, ah, false, bl,
                                                       (short)0, acc[t], false, false);
      acc[t] = __builtin_amdgcn_wmma_f32_16x16x32_bf16(false, al, false, bh,
                                                       (short)0, acc[t], false, false);
    }

    wait_async_zero();   // next slab landed in LDS
    __syncthreads();     // all waves done reading current slab
    buf ^= 1;
  }

  // Epilogue: z = acc + b1; h = silu(z); s_row = sum_col h * W2[col]  (fused GEMV)
  // C layout: lane holds col = t*16 + sub; VGPR j -> row = j + 8*khalf.
  float partial[8] = {0, 0, 0, 0, 0, 0, 0, 0};
  const float b2v = b2[0];
#pragma unroll
  for (int t = 0; t < N_TILES; ++t) {
    int col = t * 16 + sub;
    float bb = b1[col];
    float ww = W2[col];
#pragma unroll
    for (int j = 0; j < 8; ++j) {
      float z = acc[t][j] + bb;
      float h = z / (1.0f + __expf(-z));   // SiLU
      partial[j] += h * ww;
    }
  }
  // Reduce over the 16 lanes of each half-wave (butterfly stays within halves).
#pragma unroll
  for (int j = 0; j < 8; ++j) {
    partial[j] += __shfl_xor(partial[j], 8, 32);
    partial[j] += __shfl_xor(partial[j], 4, 32);
    partial[j] += __shfl_xor(partial[j], 2, 32);
    partial[j] += __shfl_xor(partial[j], 1, 32);
  }

  if (sub < 8) {
    int row = waveRow + (khalf << 3) + sub;  // lanes 0-7 -> rows 0-7; 16-23 -> rows 8-15
    if (row < E) {
      float sv = partial[0];
#pragma unroll
      for (int j = 1; j < 8; ++j) if (sub == j) sv = partial[j];
      float sc = sv + b2v;
      int node = idx_t[row];
      float vx = V_st[(size_t)row * 3 + 0];
      float vy = V_st[(size_t)row * 3 + 1];
      float vz = V_st[(size_t)row * 3 + 2];
      atomicAdd(out + (size_t)node * 3 + 0, sc * vx);
      atomicAdd(out + (size_t)node * 3 + 1, sc * vy);
      atomicAdd(out + (size_t)node * 3 + 2, sc * vz);
    }
  }
}

// ---- launcher ---------------------------------------------------------------
extern "C" void kernel_launch(void* const* d_in, const int* in_sizes, int n_in,
                              void* d_out, int out_size, void* d_ws, size_t ws_size,
                              hipStream_t stream) {
  const float* forces = (const float*)d_in[0];
  const float* V_st   = (const float*)d_in[1];
  const float* W1     = (const float*)d_in[2];
  const float* b1     = (const float*)d_in[3];
  const float* W2     = (const float*)d_in[4];
  const float* b2     = (const float*)d_in[5];
  const int*   idx_t  = (const int*)d_in[6];
  const int E = in_sizes[0] / D_MODEL;

  void* Bhi = d_ws;                                        // 128 KB bf16 hi plane
  void* Blo = (char*)d_ws + (size_t)D_MODEL * D_MODEL * 2; // 128 KB bf16 lo plane

  // Output is accumulated with atomics -> must start from zero every call.
  hipMemsetAsync(d_out, 0, (size_t)out_size * sizeof(float), stream);

  nvoh_pack_w1<<<(D_MODEL * D_MODEL) / 256, 256, 0, stream>>>(W1, Bhi, Blo);
  nvoh_main<<<(E + M_TILE - 1) / M_TILE, 256, 0, stream>>>(
      forces, V_st, b1, W2, b2, idx_t, Bhi, Blo, (float*)d_out, E);
}
